// GNNLayer_22351009809180
// MI455X (gfx1250) — matrix operand order repaired and checked by reading, not compiled, not run on hardware
//
#include <hip/hip_runtime.h>
#include <hip/hip_bf16.h>

typedef __bf16 bf16_t;
typedef __attribute__((ext_vector_type(16))) __bf16 v16bf;
typedef __attribute__((ext_vector_type(8)))  __bf16 v8bf;
typedef __attribute__((ext_vector_type(8)))  float  v8f;

#define N_NODES 50000
#define N_EDGES 800000
#define DIM     128
#define M_TILES (N_NODES / 16)   // 3125 exact tiles -> EXEC always all-ones for WMMA
#define WPB     4                // waves per block in the GEMM kernel

#define WMMA_BF16(A, B, C) \
  __builtin_amdgcn_wmma_f32_16x16x32_bf16(false, (A), false, (B), (short)0, (C), false, false)

// ---------------------------------------------------------------------------
// B-fragment packed layout: slot = ((kb*8 + nb)*32 + lane), 16 bf16 per slot.
// lane<16: col = nb*16+lane,     K = kb*32 + {0..7, 16..23}
// lane>=16: col = nb*16+lane-16, K = kb*32 + {8..15, 24..31}
// ---------------------------------------------------------------------------
__global__ void pack_w(const float* __restrict__ W, bf16_t* __restrict__ dst, int kblocks) {
  int slot = blockIdx.x * blockDim.x + threadIdx.x;
  if (slot >= kblocks * 256) return;
  int lane = slot & 31;
  int nb   = (slot >> 5) & 7;
  int kb   = slot >> 8;
  int krow = kb * 32 + ((lane & 16) ? 8 : 0);
  int col  = nb * 16 + (lane & 15);
  bf16_t* o = dst + (size_t)slot * 16;
#pragma unroll
  for (int j = 0; j < 16; ++j) o[j] = (bf16_t)W[(size_t)(krow + j + ((j & 8) ? 8 : 0)) * DIM + col];
}

__global__ void zero_ws(float* __restrict__ p, int n) {
  int i = blockIdx.x * blockDim.x + threadIdx.x;
  if (i < n) p[i] = 0.0f;
}

// ---------------------------------------------------------------------------
// Edge scatter: agg[row] += x[col] ; cnt[row] += 1.  One wave per edge,
// lane handles 4 consecutive floats (16B vector load, 4 f32 atomics).
// x and agg are L2-resident (25.6 MB each vs 192 MB L2).
// ---------------------------------------------------------------------------
__global__ void edge_aggregate(const float* __restrict__ x, const int* __restrict__ ei,
                               float* __restrict__ agg, float* __restrict__ cnt) {
  const int lane = threadIdx.x & 31;
  int w        = blockIdx.x * (blockDim.x >> 5) + (threadIdx.x >> 5);
  const int nw = gridDim.x * (blockDim.x >> 5);
  for (int e = w; e < N_EDGES; e += nw) {
    const int r = ei[e];
    const int c = ei[N_EDGES + e];
    const float4 v = *(const float4*)(x + (size_t)c * DIM + lane * 4);
    float* d = agg + (size_t)r * DIM + lane * 4;
    unsafeAtomicAdd(d + 0, v.x);
    unsafeAtomicAdd(d + 1, v.y);
    unsafeAtomicAdd(d + 2, v.z);
    unsafeAtomicAdd(d + 3, v.w);
    if (lane == 0) unsafeAtomicAdd(cnt + r, 1.0f);
  }
}

// ---------------------------------------------------------------------------
// Fragment loaders
// ---------------------------------------------------------------------------
__device__ __forceinline__ v16bf load_bfrag(const bf16_t* __restrict__ packed,
                                            int kb, int nb, int lane) {
  return *(const v16bf*)(packed + (size_t)(((kb << 3) + nb) * 32 + lane) * 16);
}

// A fragment (16x32) from an f32 row-major matrix, with per-row scale.
// lane<16: row=lane,     K = kb*32 + {0..7, 16..23}
// lane>=16: row=lane-16, K = kb*32 + {8..15, 24..31}
__device__ __forceinline__ v16bf load_afrag_f32(const float* __restrict__ row,
                                                int kb, int lane, float scale) {
  const float* p0 = row + kb * 32 + ((lane & 16) ? 8 : 0);
  const float* p1 = p0 + 16;
  v16bf a;
#pragma unroll
  for (int j = 0; j < 8; ++j) a[j] = (bf16_t)(p0[j] * scale);
#pragma unroll
  for (int j = 0; j < 8; ++j) a[j + 8] = (bf16_t)(p1[j] * scale);
  return a;
}

// A fragment from a 16x128 bf16 tile in LDS (two 16B ds loads per lane).
__device__ __forceinline__ v16bf load_afrag_lds(const bf16_t* __restrict__ tile,
                                                int kb, int lane) {
  const bf16_t* p = tile + (lane & 15) * DIM + kb * 32 + ((lane & 16) ? 8 : 0);
  v8bf lo = *(const v8bf*)p;
  v8bf hi = *(const v8bf*)(p + 16);
  return __builtin_shufflevector(lo, hi, 0, 1, 2, 3, 4, 5, 6, 7,
                                 8, 9, 10, 11, 12, 13, 14, 15);
}

// ---------------------------------------------------------------------------
// Fused per-node GEMMs. One wave owns exactly ONE 16-row tile (no outer loop,
// so weight-fragment loads cannot be hoisted/spilled):
//   S = bf16( x_tile   @ W_lin + b_lin )                       (16x128, LDS)
//   T = bf16( (agg/max(cnt,1))_tile @ W_msg + gate*b_msg )     (16x128, LDS)
//   out = relu( [S | T] @ W_upd + b_upd )                      (K = 256)
// DS ops are in-order per wave, and tiles are wave-private -> no barriers.
// nb loops kept rolled (#pragma unroll 1) to bound live B-fragments.
// ---------------------------------------------------------------------------
__global__ __launch_bounds__(WPB * 32) void fused_node_gemm(
    const float* __restrict__ x, const float* __restrict__ agg, const float* __restrict__ cnt,
    const bf16_t* __restrict__ pWlin, const bf16_t* __restrict__ pWmsg,
    const bf16_t* __restrict__ pWupd,
    const float* __restrict__ b_lin, const float* __restrict__ b_msg,
    const float* __restrict__ b_upd, float* __restrict__ out) {
  __shared__ alignas(16) bf16_t smem[WPB][2][16 * DIM];
  const int lane = threadIdx.x & 31;
  const int wv   = threadIdx.x >> 5;
  const int l15  = lane & 15;
  const int rb   = (lane & 16) ? 8 : 0;   // D-matrix row base for this lane half

  const int tile = blockIdx.x * WPB + wv; // uniform per wave
  if (tile >= M_TILES) return;            // uniform branch: EXEC stays all-ones
  const int m0 = tile << 4;

  bf16_t* S = &smem[wv][0][0];
  bf16_t* T = &smem[wv][1][0];

  // ---- stage 1a: self path -> S ----
  {
    const float* row = x + (size_t)(m0 + l15) * DIM;
    v16bf a0 = load_afrag_f32(row, 0, lane, 1.0f);
    v16bf a1 = load_afrag_f32(row, 1, lane, 1.0f);
    v16bf a2 = load_afrag_f32(row, 2, lane, 1.0f);
    v16bf a3 = load_afrag_f32(row, 3, lane, 1.0f);
#pragma unroll 1
    for (int nb = 0; nb < 8; ++nb) {
      v8f c = {};
      c = WMMA_BF16(a0, load_bfrag(pWlin, 0, nb, lane), c);
      c = WMMA_BF16(a1, load_bfrag(pWlin, 1, nb, lane), c);
      c = WMMA_BF16(a2, load_bfrag(pWlin, 2, nb, lane), c);
      c = WMMA_BF16(a3, load_bfrag(pWlin, 3, nb, lane), c);
      const float bias = b_lin[nb * 16 + l15];
#pragma unroll
      for (int v = 0; v < 8; ++v)
        S[(rb + v) * DIM + nb * 16 + l15] = (bf16_t)(c[v] + bias);
    }
  }

  // ---- stage 1b: aggregated-message path -> T ----
  {
    const float scale = 1.0f / fmaxf(cnt[m0 + l15], 1.0f);
    const float* row = agg + (size_t)(m0 + l15) * DIM;
    v16bf a0 = load_afrag_f32(row, 0, lane, scale);
    v16bf a1 = load_afrag_f32(row, 1, lane, scale);
    v16bf a2 = load_afrag_f32(row, 2, lane, scale);
    v16bf a3 = load_afrag_f32(row, 3, lane, scale);
    float gate[8];
#pragma unroll
    for (int v = 0; v < 8; ++v)
      gate[v] = (cnt[m0 + rb + v] > 0.0f) ? 1.0f : 0.0f;  // isolated nodes -> h_agg = 0
#pragma unroll 1
    for (int nb = 0; nb < 8; ++nb) {
      v8f c = {};
      c = WMMA_BF16(a0, load_bfrag(pWmsg, 0, nb, lane), c);
      c = WMMA_BF16(a1, load_bfrag(pWmsg, 1, nb, lane), c);
      c = WMMA_BF16(a2, load_bfrag(pWmsg, 2, nb, lane), c);
      c = WMMA_BF16(a3, load_bfrag(pWmsg, 3, nb, lane), c);
      const float bias = b_msg[nb * 16 + l15];
#pragma unroll
      for (int v = 0; v < 8; ++v)
        T[(rb + v) * DIM + nb * 16 + l15] = (bf16_t)(c[v] + gate[v] * bias);
    }
  }

  // ---- stage 2: concat GEMM (K=256) + bias + relu -> out ----
  {
    v16bf a[8];
#pragma unroll
    for (int kb = 0; kb < 4; ++kb) a[kb] = load_afrag_lds(S, kb, lane);
#pragma unroll
    for (int kb = 0; kb < 4; ++kb) a[4 + kb] = load_afrag_lds(T, kb, lane);
#pragma unroll 1
    for (int nb = 0; nb < 8; ++nb) {
      v8f c = {};
#pragma unroll
      for (int k = 0; k < 8; ++k)
        c = WMMA_BF16(a[k], load_bfrag(pWupd, k, nb, lane), c);
      const float bias = b_upd[nb * 16 + l15];
#pragma unroll
      for (int v = 0; v < 8; ++v)
        out[(size_t)(m0 + rb + v) * DIM + nb * 16 + l15] = fmaxf(c[v] + bias, 0.0f);
    }
  }
}

// ---------------------------------------------------------------------------
// Host side
// ---------------------------------------------------------------------------
extern "C" void kernel_launch(void* const* d_in, const int* in_sizes, int n_in,
                              void* d_out, int out_size, void* d_ws, size_t ws_size,
                              hipStream_t stream) {
  const float* x     = (const float*)d_in[0];
  const int*   ei    = (const int*)d_in[1];
  const float* W_lin = (const float*)d_in[2];
  const float* b_lin = (const float*)d_in[3];
  const float* W_msg = (const float*)d_in[4];
  const float* b_msg = (const float*)d_in[5];
  const float* W_upd = (const float*)d_in[6];
  const float* b_upd = (const float*)d_in[7];
  float* out = (float*)d_out;

  // Workspace layout (bytes):
  //   [0, 25.6M)        agg   : N_NODES*DIM f32 accumulator
  //   [25.6M, 25.8M)    cnt   : N_NODES f32 counts
  //   [25.8M, ...)      packed bf16 weights (32B-aligned)
  char*  ws    = (char*)d_ws;
  float* agg   = (float*)ws;
  float* cnt   = agg + (size_t)N_NODES * DIM;
  const size_t packOff = (size_t)N_NODES * DIM * 4 + (size_t)N_NODES * 4; // 25,800,000 (32B aligned)
  bf16_t* pWlin = (bf16_t*)(ws + packOff);                    // 4*8*32*16 bf16 = 32 KB
  bf16_t* pWmsg = pWlin + 4 * 8 * 32 * 16;
  bf16_t* pWupd = pWmsg + 4 * 8 * 32 * 16;                    // 8*8*32*16 bf16 = 64 KB

  const int nz = N_NODES * DIM + N_NODES;
  zero_ws<<<(nz + 255) / 256, 256, 0, stream>>>(agg, nz);

  pack_w<<<4, 256, 0, stream>>>(W_lin, pWlin, 4);
  pack_w<<<4, 256, 0, stream>>>(W_msg, pWmsg, 4);
  pack_w<<<8, 256, 0, stream>>>(W_upd, pWupd, 8);

  edge_aggregate<<<1024, 256, 0, stream>>>(x, ei, agg, cnt);

  fused_node_gemm<<<(M_TILES + WPB - 1) / WPB, WPB * 32, 0, stream>>>(
      x, agg, cnt, pWlin, pWmsg, pWupd, b_lin, b_msg, b_upd, out);
}
